// Param2MoEDecoderLayer_29076928594670
// MI455X (gfx1250) — matrix operand order repaired and checked by reading, not compiled, not run on hardware
//
#include <hip/hip_runtime.h>
#include <hip/hip_bf16.h>
#include <math.h>

// ---------------------------------------------------------------------------
// MI455X (gfx1250) MoE decoder layer.
// wave32; v_wmma_f32_16x16x32_bf16 for every GEMM-shaped op; sparse expert
// dispatch (token compaction) instead of the reference's dense-over-experts
// einsum. GEMMs stage tiles through LDS: A row-major (TDM tensor_load_to_lds
// when available), B transposed so WMMA fragments are contiguous ds_load_b128.
// ---------------------------------------------------------------------------

typedef __bf16 bf16_t;
typedef __attribute__((ext_vector_type(16))) __bf16 v16bf;
typedef __attribute__((ext_vector_type(8)))  __bf16 v8bf;
typedef __attribute__((ext_vector_type(8)))  float  v8f;
typedef __attribute__((ext_vector_type(4))) unsigned int v4u;
typedef __attribute__((ext_vector_type(8))) int v8i_t;
typedef __attribute__((ext_vector_type(4))) int v4i_t;

#define WMMA_BF16(a, b, c) \
  __builtin_amdgcn_wmma_f32_16x16x32_bf16(false, (a), false, (b), (short)0, (c), false, false)

#if defined(__has_builtin)
#if __has_builtin(__builtin_amdgcn_tensor_load_to_lds) && \
    __has_builtin(__builtin_amdgcn_s_wait_tensorcnt)
#define HAVE_TDM 1
#endif
#endif

static constexpr int cNH   = 16;
static constexpr int cNKV  = 4;
static constexpr int cDH   = 64;
static constexpr int cE    = 64;
static constexpr int cNG   = 8;
static constexpr int cTKG  = 4;
static constexpr int cTOPK = 8;
static constexpr int cH    = 1024;
static constexpr int cI    = 256;
static constexpr int cIS   = 512;

#ifdef HAVE_TDM
// Build a CDNA5 Tensor DMA Descriptor (ISA cdna5 §8) for a 2-D bf16 tile
// (rows x cols, row stride = stride_elems) and issue TENSOR_LOAD_TO_LDS.
__device__ __forceinline__ void tdm_load_2d(unsigned lds_addr, const void* gptr,
                                            int rows, int cols, int stride_elems) {
  unsigned long long ga = (unsigned long long)gptr;
  v4u g0;
  g0[0] = 1u;                                           // count=1, user D#
  g0[1] = lds_addr;                                     // lds_addr[31:0]
  g0[2] = (unsigned)(ga & 0xffffffffu);                 // global_addr[31:0]
  g0[3] = (unsigned)((ga >> 32) & 0x01ffffffu) | (2u << 30);  // addr[56:32], type=2
  unsigned td0 = (unsigned)stride_elems;                // tensor_dim0 (elems)
  unsigned td1 = 0x7fffffffu;                           // tensor_dim1 (huge)
  unsigned long long st0 = (unsigned long long)stride_elems;
  v8i_t g1;
  g1[0] = (int)(1u << 16);                              // data_size=1 (2 bytes)
  g1[1] = (int)((td0 & 0xffffu) << 16);                 // tensor_dim0[15:0] @48
  g1[2] = (int)((td0 >> 16) | ((td1 & 0xffffu) << 16)); // dim0 hi | dim1 lo
  g1[3] = (int)((td1 >> 16) | ((unsigned)cols << 16));  // dim1 hi | tile_dim0
  g1[4] = rows;                                         // tile_dim1 (tile_dim2=0)
  g1[5] = (int)(st0 & 0xffffffffu);                     // tensor_dim0_stride lo
  g1[6] = (int)((st0 >> 32) & 0xffffu);                 // stride hi
  g1[7] = 0;
  v4i_t g2 = {0, 0, 0, 0};
  v4i_t g3 = {0, 0, 0, 0};
#if __clang_major__ >= 23
  v8i_t g4 = {0, 0, 0, 0, 0, 0, 0, 0};
  __builtin_amdgcn_tensor_load_to_lds(g0, g1, g2, g3, g4, 0);
#else
  __builtin_amdgcn_tensor_load_to_lds(g0, g1, g2, g3, 0);
#endif
  __builtin_amdgcn_s_wait_tensorcnt(0);
}
#endif

// --------------------------- WMMA fragment loaders -------------------------
// A 16x32 bf16: lanes 0-15 row M=lane, K=0..15; lanes
// 16-31 row M=lane-16, K=16..31. Two contiguous 16B loads.
__device__ __forceinline__ v16bf load_a_frag(const bf16_t* __restrict__ A, int lda,
                                             int m0, int k0, int lane) {
  int m  = m0 + (lane & 15);
  int kb = (lane < 16) ? 0 : 16;
  const bf16_t* p = A + (size_t)m * lda + k0 + kb;
  v8bf lo = *(const v8bf*)p;
  v8bf hi = *(const v8bf*)(p + 8);
  v16bf r;
#pragma unroll
  for (int i = 0; i < 8; i++) { r[i] = lo[i]; r[8 + i] = hi[i]; }
  return r;
}

__device__ __forceinline__ v16bf load_a_gather(const bf16_t* __restrict__ A, int lda,
                                               const int* toks, int k0, int lane) {
  int r  = lane & 15;
  int kb = (lane < 16) ? 0 : 16;
  const bf16_t* p = A + (size_t)toks[r] * lda + k0 + kb;
  v8bf lo = *(const v8bf*)p;
  v8bf hi = *(const v8bf*)(p + 8);
  v16bf out;
#pragma unroll
  for (int i = 0; i < 8; i++) { out[i] = lo[i]; out[8 + i] = hi[i]; }
  return out;
}

// B fragment from a K-contiguous layout S[n][k] (transposed LDS tile or K^T
// in global): per lane 16 contiguous bf16 -> two 16B loads.
__device__ __forceinline__ v16bf load_bT_frag(const bf16_t* __restrict__ S, int ldk,
                                              int n0, int k0, int lane) {
  int n  = n0 + (lane & 15);
  int kh = (lane < 16) ? 0 : 16;
  const bf16_t* p = S + (size_t)n * ldk + k0 + kh;
  v8bf lo = *(const v8bf*)p;
  v8bf hi = *(const v8bf*)(p + 8);
  v16bf r;
#pragma unroll
  for (int i = 0; i < 8; i++) { r[i] = lo[i]; r[8 + i] = hi[i]; }
  return r;
}

// B fragment with strided access (only used for small V tiles in attention).
__device__ __forceinline__ v16bf load_b_frag(const bf16_t* __restrict__ B, int ldb,
                                             int k0, int n0, int lane) {
  int n  = n0 + (lane & 15);
  int kh = (lane < 16) ? 0 : 16;
  const bf16_t* p = B + (size_t)(k0 + kh) * ldb + n;
  v16bf r;
#pragma unroll
  for (int v = 0; v < 8; v++) {
    r[2 * v]     = p[(size_t)(2 * v) * ldb];
    r[2 * v + 1] = p[(size_t)(2 * v + 1) * ldb];
  }
  return r;
}

// ------------------------------ small kernels ------------------------------
__global__ void cvt_bf16_kernel(const float* __restrict__ in, bf16_t* __restrict__ out, size_t n) {
  size_t i = (size_t)blockIdx.x * 256 + threadIdx.x;
  size_t stride = (size_t)gridDim.x * 256;
  for (; i < n; i += stride) out[i] = (bf16_t)in[i];
}

__global__ void gate_transpose_kernel(const float* __restrict__ in, bf16_t* __restrict__ out) {
  int i = blockIdx.x * 256 + threadIdx.x;   // over E*H
  if (i < cE * cH) { int e = i / cH, h = i % cH; out[h * cE + e] = (bf16_t)in[i]; }
}

__global__ __launch_bounds__(256) void rmsnorm_bf16_kernel(const float* __restrict__ x,
                                                           const float* __restrict__ w,
                                                           bf16_t* __restrict__ out, int H) {
  __shared__ float red[256];
  int t = blockIdx.x;
  const float* xr = x + (size_t)t * H;
  float s = 0.f;
  for (int i = threadIdx.x; i < H; i += 256) { float v = xr[i]; s += v * v; }
  red[threadIdx.x] = s;
  __syncthreads();
  for (int st = 128; st > 0; st >>= 1) {
    if ((int)threadIdx.x < st) red[threadIdx.x] += red[threadIdx.x + st];
    __syncthreads();
  }
  float rs = rsqrtf(red[0] / (float)H + 1e-6f);
  for (int i = threadIdx.x; i < H; i += 256)
    out[(size_t)t * H + i] = (bf16_t)(xr[i] * rs * w[i]);
}

__global__ __launch_bounds__(256) void rope_pack_kernel(const float* __restrict__ qkv,
                                                        const int* __restrict__ pos,
                                                        bf16_t* __restrict__ Qo,
                                                        bf16_t* __restrict__ Ko,
                                                        bf16_t* __restrict__ Vo) {
  int t = blockIdx.x;
  float p = (float)pos[t];
  const float* row = qkv + (size_t)t * (cNH + 2 * cNKV) * cDH;
  for (int w = threadIdx.x; w < (cNH + cNKV) * 32; w += 256) {
    int hh = w / 32, j = w % 32;
    float inv = __expf(-((float)(2 * j) / 64.f) * __logf(10000.f));
    float f = p * inv, sn, cs;
    __sincosf(f, &sn, &cs);
    const float* b = row + hh * cDH;
    float x1 = b[j], x2 = b[j + 32];
    float o1 = x1 * cs - x2 * sn;
    float o2 = x2 * cs + x1 * sn;
    if (hh < cNH) {
      bf16_t* q = Qo + (size_t)t * cNH * cDH + hh * cDH;
      q[j] = (bf16_t)(o1 * 0.125f);        // fold D^-0.5 into Q
      q[j + 32] = (bf16_t)(o2 * 0.125f);
    } else {
      int kh = hh - cNH;
      bf16_t* k = Ko + (size_t)t * cNKV * cDH + kh * cDH;
      k[j] = (bf16_t)o1;
      k[j + 32] = (bf16_t)o2;
    }
  }
  const float* vsrc = row + (cNH + cNKV) * cDH;
  for (int i = threadIdx.x; i < cNKV * cDH; i += 256)
    Vo[(size_t)t * cNKV * cDH + i] = (bf16_t)vsrc[i];
}

__global__ void swiglu_kernel(const float* __restrict__ gu, bf16_t* __restrict__ out, int T) {
  int i = blockIdx.x * 256 + threadIdx.x;
  if (i < T * cIS) {
    int t = i / cIS, j = i % cIS;
    float g = gu[(size_t)t * 2 * cIS + j];
    float u = gu[(size_t)t * 2 * cIS + cIS + j];
    out[i] = (bf16_t)(g / (1.f + __expf(-g)) * u);
  }
}

// ------------------------------ generic GEMM -------------------------------
// C[M,N] = A x B (+resid). Block tile 128x64 (8 waves, each 16x64). Per
// 32-deep K step: A tile staged to LDS (TDM if available), B tile staged
// transposed so every fragment load is 2x ds_load_b128.
static constexpr int GBM = 128, GBN = 64, GKC = 32;

__global__ __launch_bounds__(256) void gemm_bf16_kernel(const bf16_t* __restrict__ A,
                                                        const bf16_t* __restrict__ B,
                                                        float* __restrict__ C,
                                                        const float* __restrict__ resid,
                                                        int M, int N, int K) {
  __shared__ alignas(16) bf16_t sA[GBM * GKC];    // [128][32] row-major, 8 KB
  __shared__ alignas(16) bf16_t sBt[GBN * GKC];   // [64][32] transposed, 4 KB
  const int tid = threadIdx.x;
  const int lane = tid & 31, wid = tid >> 5;
  const int NT = N / GBN;
  const int mt = blockIdx.x / NT, nt = blockIdx.x % NT;
  const int m0 = mt * GBM, n0 = nt * GBN;
  v8f c0 = {}, c1 = {}, c2 = {}, c3 = {};
  for (int k0 = 0; k0 < K; k0 += GKC) {
    // ---- stage A (128x32) ----
#ifdef HAVE_TDM
    if (wid == 0)
      tdm_load_2d((unsigned)(size_t)(void*)sA,
                  (const void*)(A + (size_t)m0 * K + k0), GBM, GKC, K);
#else
    {
      int row = tid >> 1, half = (tid & 1) * 16;
      const v8bf* src = (const v8bf*)(A + (size_t)(m0 + row) * K + k0 + half);
      v8bf* dst = (v8bf*)(sA + row * GKC + half);
      dst[0] = src[0];
      dst[1] = src[1];
    }
#endif
    // ---- stage B transposed: B[32][64] -> sBt[64][32] ----
    {
      int k = tid >> 3;             // 0..31
      int ng = (tid & 7) * 8;       // 0..56
      v8bf bv = *(const v8bf*)(B + (size_t)(k0 + k) * N + n0 + ng);
#pragma unroll
      for (int i = 0; i < 8; i++) sBt[(ng + i) * GKC + k] = bv[i];
    }
    __syncthreads();
    v16bf a = load_a_frag(sA, GKC, wid * 16, 0, lane);
    c0 = WMMA_BF16(a, load_bT_frag(sBt, GKC, 0,  0, lane), c0);
    c1 = WMMA_BF16(a, load_bT_frag(sBt, GKC, 16, 0, lane), c1);
    c2 = WMMA_BF16(a, load_bT_frag(sBt, GKC, 32, 0, lane), c2);
    c3 = WMMA_BF16(a, load_bT_frag(sBt, GKC, 48, 0, lane), c3);
    __syncthreads();
  }
  int n = lane & 15, mb = (lane < 16) ? 0 : 8;
#pragma unroll
  for (int r = 0; r < 8; r++) {
    size_t row = (size_t)(m0 + wid * 16 + mb + r) * N;
    float r0 = resid ? resid[row + n0 + n]      : 0.f;
    float r1 = resid ? resid[row + n0 + 16 + n] : 0.f;
    float r2 = resid ? resid[row + n0 + 32 + n] : 0.f;
    float r3 = resid ? resid[row + n0 + 48 + n] : 0.f;
    C[row + n0 + n]      = c0[r] + r0;
    C[row + n0 + 16 + n] = c1[r] + r1;
    C[row + n0 + 32 + n] = c2[r] + r2;
    C[row + n0 + 48 + n] = c3[r] + r3;
  }
}

// --------------------------- flash attention (GQA) -------------------------
__global__ __launch_bounds__(32) void flash_attn_kernel(const bf16_t* __restrict__ Q,
                                                        const bf16_t* __restrict__ Kb,
                                                        const bf16_t* __restrict__ Vb,
                                                        bf16_t* __restrict__ O) {
  __shared__ alignas(16) bf16_t pL[16 * 32];
  const int lane = threadIdx.x & 31;
  const int h    = blockIdx.y;
  const int q0   = blockIdx.x * 16;
  const int kvh  = h / (cNH / cNKV);
  const int n    = lane & 15;
  const int mb   = (lane < 16) ? 0 : 8;

  v16bf aQ0 = load_a_frag(Q + h * cDH, cNH * cDH, q0, 0, lane);
  v16bf aQ1 = load_a_frag(Q + h * cDH, cNH * cDH, q0, 32, lane);

  v8f acc0 = {}, acc1 = {}, acc2 = {}, acc3 = {};
  float m[8], l[8];
#pragma unroll
  for (int r = 0; r < 8; r++) { m[r] = -1e30f; l[r] = 0.f; }

  const bf16_t* Kc = Kb + kvh * cDH;
  const bf16_t* Vc = Vb + kvh * cDH;
  const int nc = (q0 + 16 + 31) >> 5;
  for (int c = 0; c < nc; c++) {
    int key0 = c * 32;
    v8f s0 = {}, s1 = {};
    s0 = WMMA_BF16(aQ0, load_bT_frag(Kc, cNKV * cDH, key0,      0,  lane), s0);
    s0 = WMMA_BF16(aQ1, load_bT_frag(Kc, cNKV * cDH, key0,      32, lane), s0);
    s1 = WMMA_BF16(aQ0, load_bT_frag(Kc, cNKV * cDH, key0 + 16, 0,  lane), s1);
    s1 = WMMA_BF16(aQ1, load_bT_frag(Kc, cNKV * cDH, key0 + 16, 32, lane), s1);
#pragma unroll
    for (int r = 0; r < 8; r++) {
      int row = q0 + mb + r;
      float v0 = ((key0 + n)      <= row) ? s0[r] : -1e30f;
      float v1 = ((key0 + 16 + n) <= row) ? s1[r] : -1e30f;
      float mx = fmaxf(v0, v1);
      mx = fmaxf(mx, __shfl_xor(mx, 1, 16));
      mx = fmaxf(mx, __shfl_xor(mx, 2, 16));
      mx = fmaxf(mx, __shfl_xor(mx, 4, 16));
      mx = fmaxf(mx, __shfl_xor(mx, 8, 16));
      float mn = fmaxf(m[r], mx);
      float cf = __expf(m[r] - mn);
      m[r] = mn;
      float p0 = __expf(v0 - mn);
      float p1 = __expf(v1 - mn);
      float rs = p0 + p1;
      rs += __shfl_xor(rs, 1, 16);
      rs += __shfl_xor(rs, 2, 16);
      rs += __shfl_xor(rs, 4, 16);
      rs += __shfl_xor(rs, 8, 16);
      l[r] = l[r] * cf + rs;
      acc0[r] *= cf; acc1[r] *= cf; acc2[r] *= cf; acc3[r] *= cf;
      pL[(mb + r) * 32 + n]      = (bf16_t)p0;
      pL[(mb + r) * 32 + 16 + n] = (bf16_t)p1;
    }
    __syncthreads();
    v16bf aP = load_a_frag(pL, 32, 0, 0, lane);
    acc0 = WMMA_BF16(aP, load_b_frag(Vc, cNKV * cDH, key0, 0,  lane), acc0);
    acc1 = WMMA_BF16(aP, load_b_frag(Vc, cNKV * cDH, key0, 16, lane), acc1);
    acc2 = WMMA_BF16(aP, load_b_frag(Vc, cNKV * cDH, key0, 32, lane), acc2);
    acc3 = WMMA_BF16(aP, load_b_frag(Vc, cNKV * cDH, key0, 48, lane), acc3);
    __syncthreads();
  }
#pragma unroll
  for (int r = 0; r < 8; r++) {
    float inv = 1.0f / l[r];
    bf16_t* o = O + (size_t)(q0 + mb + r) * (cNH * cDH) + h * cDH;
    o[n]      = (bf16_t)(acc0[r] * inv);
    o[16 + n] = (bf16_t)(acc1[r] * inv);
    o[32 + n] = (bf16_t)(acc2[r] * inv);
    o[48 + n] = (bf16_t)(acc3[r] * inv);
  }
}

// ------------------------------- MoE routing -------------------------------
__global__ __launch_bounds__(64) void route_kernel(const float* __restrict__ logits,
                                                   const float* __restrict__ ebias,
                                                   float* __restrict__ combine) {
  __shared__ float ss[cE], sc[cE];
  int t = blockIdx.x, e = threadIdx.x;
  float s = 1.f / (1.f + __expf(-logits[(size_t)t * cE + e]));
  ss[e] = s;
  sc[e] = s + ebias[e];
  combine[(size_t)t * cE + e] = 0.f;
  __syncthreads();
  if (e == 0) {
    float gsc[cNG];
    for (int g = 0; g < cNG; g++) {
      float m1 = -3e38f, m2 = -3e38f;
      for (int j = 0; j < cE / cNG; j++) {
        float v = sc[g * (cE / cNG) + j];
        if (v > m1) { m2 = m1; m1 = v; } else if (v > m2) { m2 = v; }
      }
      gsc[g] = m1 + m2;
    }
    bool gsel[cNG];
    for (int g = 0; g < cNG; g++) gsel[g] = false;
    for (int it = 0; it < cTKG; it++) {
      int bi = 0; float bv = -3e38f;
      for (int g = 0; g < cNG; g++)
        if (!gsel[g] && gsc[g] > bv) { bv = gsc[g]; bi = g; }
      gsel[bi] = true;
    }
    bool tk[cE];
    for (int i = 0; i < cE; i++) tk[i] = false;
    int ids[cTOPK]; float wv[cTOPK]; float wsum = 0.f;
    for (int it = 0; it < cTOPK; it++) {
      int bi = 0; float bv = -3e38f;
      for (int e2 = 0; e2 < cE; e2++) {
        if (tk[e2] || !gsel[e2 >> 3]) continue;
        if (sc[e2] > bv) { bv = sc[e2]; bi = e2; }
      }
      tk[bi] = true; ids[it] = bi; wv[it] = ss[bi]; wsum += ss[bi];
    }
    float inv = 2.5f / (wsum + 1e-20f);
    for (int it = 0; it < cTOPK; it++)
      combine[(size_t)t * cE + ids[it]] = wv[it] * inv;
  }
}

__global__ __launch_bounds__(64) void compact_kernel(const float* __restrict__ combine,
                                                     int* __restrict__ counts,
                                                     int* __restrict__ off,
                                                     int* __restrict__ idx,
                                                     float* __restrict__ wt, int T) {
  int e = threadIdx.x;
  int c = 0;
  for (int t = 0; t < T; t++)
    if (combine[(size_t)t * cE + e] != 0.f) c++;
  counts[e] = c;
  __syncthreads();
  if (e == 0) { int a = 0; for (int i = 0; i < cE; i++) { off[i] = a; a += counts[i]; } }
  __syncthreads();
  int o = off[e]; c = 0;
  for (int t = 0; t < T; t++) {
    float w = combine[(size_t)t * cE + e];
    if (w != 0.f) { idx[o + c] = t; wt[o + c] = w; c++; }
  }
}

// ---------------------- routed experts: gate/up + SwiGLU -------------------
__global__ __launch_bounds__(256) void routed_gateup_kernel(const bf16_t* __restrict__ H2,
                                                            const bf16_t* __restrict__ Wg,
                                                            const bf16_t* __restrict__ Wu,
                                                            const int* __restrict__ counts,
                                                            const int* __restrict__ off,
                                                            const int* __restrict__ idx,
                                                            bf16_t* __restrict__ Act) {
  int e = blockIdx.y, rt = blockIdx.x;
  int cnt = counts[e];
  if (rt * 16 >= cnt) return;
  int base = off[e] + rt * 16;
  __shared__ int sTok[16];
  __shared__ alignas(16) float  sG[16 * cI];       // 16 KB
  __shared__ alignas(16) float  sU[16 * cI];       // 16 KB
  __shared__ alignas(16) bf16_t sBtG[cI * 32];     // [256][32] 16 KB
  __shared__ alignas(16) bf16_t sBtU[cI * 32];     // 16 KB
  const int tid = threadIdx.x;
  if (tid < 16) {
    int rr = rt * 16 + tid;
    sTok[tid] = idx[(rr < cnt) ? (off[e] + rr) : off[e]];
  }
  __syncthreads();
  int lane = tid & 31, wid = tid >> 5;
  const bf16_t* Bg = Wg + (size_t)e * cH * cI;
  const bf16_t* Bu = Wu + (size_t)e * cH * cI;
  int n0 = (wid & 3) * 64;
  v8f c0 = {}, c1 = {}, c2 = {}, c3 = {};
  for (int k0 = 0; k0 < cH; k0 += 32) {
    {   // stage both 32x256 weight tiles transposed
      int k = tid >> 3;             // 0..31
      int ng = (tid & 7) * 32;      // 0..224
      const v8bf* pg = (const v8bf*)(Bg + (size_t)(k0 + k) * cI + ng);
      const v8bf* pu = (const v8bf*)(Bu + (size_t)(k0 + k) * cI + ng);
#pragma unroll
      for (int q = 0; q < 4; q++) {
        v8bf g8 = pg[q], u8 = pu[q];
#pragma unroll
        for (int i = 0; i < 8; i++) {
          sBtG[(ng + q * 8 + i) * 32 + k] = g8[i];
          sBtU[(ng + q * 8 + i) * 32 + k] = u8[i];
        }
      }
    }
    __syncthreads();
    v16bf a = load_a_gather(H2, cH, sTok, k0, lane);
    const bf16_t* Bt = (wid < 4) ? sBtG : sBtU;
    c0 = WMMA_BF16(a, load_bT_frag(Bt, 32, n0,      0, lane), c0);
    c1 = WMMA_BF16(a, load_bT_frag(Bt, 32, n0 + 16, 0, lane), c1);
    c2 = WMMA_BF16(a, load_bT_frag(Bt, 32, n0 + 32, 0, lane), c2);
    c3 = WMMA_BF16(a, load_bT_frag(Bt, 32, n0 + 48, 0, lane), c3);
    __syncthreads();
  }
  int n = lane & 15, mb = (lane < 16) ? 0 : 8;
  float* S = (wid < 4) ? sG : sU;
#pragma unroll
  for (int r = 0; r < 8; r++) {
    S[(mb + r) * cI + n0 + n]      = c0[r];
    S[(mb + r) * cI + n0 + 16 + n] = c1[r];
    S[(mb + r) * cI + n0 + 32 + n] = c2[r];
    S[(mb + r) * cI + n0 + 48 + n] = c3[r];
  }
  __syncthreads();
  for (int i = tid; i < 16 * cI; i += 256) {
    int r = i >> 8;
    if (rt * 16 + r < cnt) {
      float g = sG[i], u = sU[i];
      Act[(size_t)(base + r) * cI + (i & (cI - 1))] = (bf16_t)(g / (1.f + __expf(-g)) * u);
    }
  }
}

// ------------------- routed experts: down-proj + scatter -------------------
__global__ __launch_bounds__(256) void routed_down_kernel(const bf16_t* __restrict__ Act,
                                                          const bf16_t* __restrict__ Wd,
                                                          const int* __restrict__ counts,
                                                          const int* __restrict__ off,
                                                          const int* __restrict__ idx,
                                                          const float* __restrict__ wt,
                                                          float* __restrict__ Y) {
  int e = blockIdx.y, rt = blockIdx.x;
  int cnt = counts[e];
  if (rt * 16 >= cnt) return;
  int base = off[e] + rt * 16;
  __shared__ int sTok[16];
  __shared__ float sW[16];
  __shared__ alignas(16) bf16_t sBt[cH * 32];     // [1024][32] transposed, 64 KB
  const int tid = threadIdx.x;
  if (tid < 16) {
    int rr = rt * 16 + tid;
    bool v = rr < cnt;
    sTok[tid] = v ? idx[off[e] + rr] : 0;
    sW[tid]   = v ? wt[off[e] + rr] : 0.f;
  }
  __syncthreads();
  int lane = tid & 31, wid = tid >> 5;
  const bf16_t* A = Act + (size_t)base * cI;
  const bf16_t* B = Wd + (size_t)e * cI * cH;
  int n = lane & 15, mb = (lane < 16) ? 0 : 8;
  v8f acc[2][4];
#pragma unroll
  for (int hh = 0; hh < 2; hh++)
#pragma unroll
    for (int j = 0; j < 4; j++) acc[hh][j] = (v8f){};
  for (int k0 = 0; k0 < cI; k0 += 32) {
    {   // stage 32x1024 weight tile transposed
      int k = tid >> 3;             // 0..31
      int ng = (tid & 7) * 128;     // 0..896
#pragma unroll
      for (int q = 0; q < 16; q++) {
        v8bf b8 = *(const v8bf*)(B + (size_t)(k0 + k) * cH + ng + q * 8);
#pragma unroll
        for (int i = 0; i < 8; i++) sBt[(ng + q * 8 + i) * 32 + k] = b8[i];
      }
    }
    __syncthreads();
    v16bf a = load_a_frag(A, cI, 0, k0, lane);
#pragma unroll
    for (int hh = 0; hh < 2; hh++) {
      int n0 = (wid + 8 * hh) * 64;
      acc[hh][0] = WMMA_BF16(a, load_bT_frag(sBt, 32, n0,      0, lane), acc[hh][0]);
      acc[hh][1] = WMMA_BF16(a, load_bT_frag(sBt, 32, n0 + 16, 0, lane), acc[hh][1]);
      acc[hh][2] = WMMA_BF16(a, load_bT_frag(sBt, 32, n0 + 32, 0, lane), acc[hh][2]);
      acc[hh][3] = WMMA_BF16(a, load_bT_frag(sBt, 32, n0 + 48, 0, lane), acc[hh][3]);
    }
    __syncthreads();
  }
#pragma unroll
  for (int hh = 0; hh < 2; hh++) {
    int n0 = (wid + 8 * hh) * 64;
#pragma unroll
    for (int r = 0; r < 8; r++) {
      int row = mb + r;
      if (rt * 16 + row < cnt) {
        float wgt = sW[row];
        float* y = Y + (size_t)sTok[row] * cH;
        atomicAdd(y + n0 + n,      acc[hh][0][r] * wgt);
        atomicAdd(y + n0 + 16 + n, acc[hh][1][r] * wgt);
        atomicAdd(y + n0 + 32 + n, acc[hh][2][r] * wgt);
        atomicAdd(y + n0 + 48 + n, acc[hh][3][r] * wgt);
      }
    }
  }
}

// --------------------------------- launch ----------------------------------
extern "C" void kernel_launch(void* const* d_in, const int* in_sizes, int n_in,
                              void* d_out, int out_size, void* d_ws, size_t ws_size,
                              hipStream_t stream) {
  (void)n_in; (void)out_size; (void)ws_size;
  const float* x     = (const float*)d_in[0];
  const int*   pos   = (const int*)d_in[1];
  const float* ln1   = (const float*)d_in[2];
  const float* wqkv  = (const float*)d_in[3];
  const float* wo    = (const float*)d_in[4];
  const float* ln2   = (const float*)d_in[5];
  const float* gatew = (const float*)d_in[6];
  const float* ebias = (const float*)d_in[7];
  const float* weg   = (const float*)d_in[8];
  const float* weu   = (const float*)d_in[9];
  const float* wed   = (const float*)d_in[10];
  const float* wsgu  = (const float*)d_in[11];
  const float* wsd   = (const float*)d_in[12];
  float* out = (float*)d_out;

  const int T = in_sizes[0] / cH;
  const int QKVN = (cNH + 2 * cNKV) * cDH;   // 1536

  unsigned char* wp = (unsigned char*)d_ws;
  auto carve = [&](size_t bytes) -> void* {
    void* p = (void*)wp;
    wp += (bytes + 255) & ~(size_t)255;
    return p;
  };
  bf16_t* wqkv_bf = (bf16_t*)carve((size_t)cH * QKVN * 2);
  bf16_t* wo_bf   = (bf16_t*)carve((size_t)cH * cH * 2);
  bf16_t* weg_bf  = (bf16_t*)carve((size_t)cE * cH * cI * 2);
  bf16_t* weu_bf  = (bf16_t*)carve((size_t)cE * cH * cI * 2);
  bf16_t* wed_bf  = (bf16_t*)carve((size_t)cE * cI * cH * 2);
  bf16_t* wsgu_bf = (bf16_t*)carve((size_t)cH * 2 * cIS * 2);
  bf16_t* wsd_bf  = (bf16_t*)carve((size_t)cIS * cH * 2);
  bf16_t* gwT_bf  = (bf16_t*)carve((size_t)cH * cE * 2);
  bf16_t* hbf     = (bf16_t*)carve((size_t)T * cH * 2);
  float*  qkv     = (float*)carve((size_t)T * QKVN * 4);
  bf16_t* qbf     = (bf16_t*)carve((size_t)T * cNH * cDH * 2);
  bf16_t* kbf     = (bf16_t*)carve((size_t)T * cNKV * cDH * 2);
  bf16_t* vbf     = (bf16_t*)carve((size_t)T * cNKV * cDH * 2);
  bf16_t* attnbf  = (bf16_t*)carve((size_t)T * cNH * cDH * 2);
  float*  resid   = (float*)carve((size_t)T * cH * 4);
  bf16_t* h2bf    = (bf16_t*)carve((size_t)T * cH * 2);
  float*  logits  = (float*)carve((size_t)T * cE * 4);
  float*  combine = (float*)carve((size_t)T * cE * 4);
  int*    counts  = (int*)carve(cE * 4);
  int*    offs    = (int*)carve(cE * 4);
  int*    idxl    = (int*)carve((size_t)T * cTOPK * 4);
  float*  wtl     = (float*)carve((size_t)T * cTOPK * 4);
  bf16_t* act_bf  = (bf16_t*)carve((size_t)T * cTOPK * cI * 2);
  float*  gu      = (float*)carve((size_t)T * 2 * cIS * 4);   // also pads act overreads
  bf16_t* sact_bf = (bf16_t*)carve((size_t)T * cIS * 2);

  auto cdiv = [](long a, long b) { return (int)((a + b - 1) / b); };
  auto cvt = [&](const float* in, bf16_t* o, size_t n) {
    int blocks = cdiv((long)n, 256); if (blocks > 65536) blocks = 65536;
    cvt_bf16_kernel<<<blocks, 256, 0, stream>>>(in, o, n);
  };
  auto gemm = [&](const bf16_t* A, const bf16_t* B, float* C, const float* R,
                  int M, int N, int K) {
    int blocks = (M / GBM) * (N / GBN);
    gemm_bf16_kernel<<<blocks, 256, 0, stream>>>(A, B, C, R, M, N, K);
  };

  cvt(wqkv, wqkv_bf, (size_t)cH * QKVN);
  cvt(wo,   wo_bf,   (size_t)cH * cH);
  cvt(weg,  weg_bf,  (size_t)cE * cH * cI);
  cvt(weu,  weu_bf,  (size_t)cE * cH * cI);
  cvt(wed,  wed_bf,  (size_t)cE * cI * cH);
  cvt(wsgu, wsgu_bf, (size_t)cH * 2 * cIS);
  cvt(wsd,  wsd_bf,  (size_t)cIS * cH);
  gate_transpose_kernel<<<cdiv(cE * cH, 256), 256, 0, stream>>>(gatew, gwT_bf);

  // ---- attention ----
  rmsnorm_bf16_kernel<<<T, 256, 0, stream>>>(x, ln1, hbf, cH);
  gemm(hbf, wqkv_bf, qkv, nullptr, T, QKVN, cH);
  rope_pack_kernel<<<T, 256, 0, stream>>>(qkv, pos, qbf, kbf, vbf);
  {
    dim3 g(T / 16, cNH);
    flash_attn_kernel<<<g, 32, 0, stream>>>(qbf, kbf, vbf, attnbf);
  }
  gemm(attnbf, wo_bf, resid, x, T, cH, cH);

  // ---- MoE ----
  rmsnorm_bf16_kernel<<<T, 256, 0, stream>>>(resid, ln2, h2bf, cH);
  gemm(h2bf, gwT_bf, logits, nullptr, T, cE, cH);
  route_kernel<<<T, 64, 0, stream>>>(logits, ebias, combine);
  compact_kernel<<<1, 64, 0, stream>>>(combine, counts, offs, idxl, wtl, T);
  {
    dim3 g(T / 16, cE);
    routed_gateup_kernel<<<g, 256, 0, stream>>>(h2bf, weg_bf, weu_bf, counts, offs, idxl, act_bf);
  }
  gemm(h2bf, wsgu_bf, gu, nullptr, T, 2 * cIS, cH);
  swiglu_kernel<<<cdiv((long)T * cIS, 256), 256, 0, stream>>>(gu, sact_bf, T);
  gemm(sact_bf, wsd_bf, out, resid, T, cH, cIS);
  {
    dim3 g(T / 16, cE);
    routed_down_kernel<<<g, 256, 0, stream>>>(act_bf, wed_bf, counts, offs, idxl, wtl, out);
  }
}
